// GRUD_45045617000563
// MI455X (gfx1250) — compile-verified, matrix operand order checked by reading, not run on hardware
//
#include <hip/hip_runtime.h>

typedef __attribute__((ext_vector_type(16))) _Float16     v16h;
typedef __attribute__((ext_vector_type(8)))  float        v8f;
typedef __attribute__((ext_vector_type(4)))  unsigned int v4u;
typedef __attribute__((ext_vector_type(8)))  int          v8i;
typedef __attribute__((ext_vector_type(4)))  int          v4i;

#define NB   512   // batch
#define NT   128   // time/features per step-vector
#define NS   128   // scan steps (== D)
#define NH   256   // hidden
#define DIN  512   // 2*T + H
#define BM   16    // batch rows per block (one WMMA M-tile)
#define XROW (3 * NT)          // 384 interleaved m/x/d entries per batch row
#define XSLICE (BM * XROW)     // 6144 floats staged per step per block

// ---------------------------------------------------------------------------
// Kernel 0: convert GEMM weights f32 -> f16 (row-major (N,K) layout kept, so
// a B-fragment for wmma is a contiguous 32B run of one weight row).
// ---------------------------------------------------------------------------
__global__ __launch_bounds__(256) void grud_convert(
    const float* __restrict__ Wz, const float* __restrict__ Wr,
    const float* __restrict__ Wh, const float* __restrict__ Wgh,
    _Float16* __restrict__ wz, _Float16* __restrict__ wr,
    _Float16* __restrict__ wh, _Float16* __restrict__ wgh) {
  int i = blockIdx.x * 256 + threadIdx.x;
  if (i < NH * DIN) {
    wz[i] = (_Float16)Wz[i];
    wr[i] = (_Float16)Wr[i];
    wh[i] = (_Float16)Wh[i];
  }
  if (i < NH * NT) wgh[i] = (_Float16)Wgh[i];
}

// ---------------------------------------------------------------------------
// TDM issue: DMA one step's X slice (tile 1 x 384 x 16, 4B elements) to LDS.
//   element (r, q): global = X[((b0+r)*384 + q)*128 + s], lds = sX[r*384 + q]
// Descriptor per CDNA5 ISA.
// ---------------------------------------------------------------------------
__device__ __forceinline__ void tdm_issue_xslice(const float* X, int b0, int s,
                                                 unsigned lds_off) {
  unsigned long long ga = (unsigned long long)(const void*)X +
                          ((unsigned long long)b0 * XROW * NS + (unsigned)s) * 4ull;
  v4u g0 = { 1u,                                   // count=1, no gather
             lds_off,                              // lds_addr
             (unsigned)(ga & 0xffffffffull),       // global_addr[31:0]
             (unsigned)((ga >> 32) & 0x1ffffffull) // global_addr[56:32]
                 | (2u << 30) };                   // type=2 (image)
  v8i g1 = { (int)(2u << 16),                 // data_size=2 (4B), mask=0
             (int)((unsigned)NS << 16),       // tensor_dim0 = 128
             (int)((unsigned)XROW << 16),     // dim0 hi=0 | tensor_dim1 = 384
             (int)(1u << 16),                 // dim1 hi=0 | tile_dim0 = 1
             (int)((unsigned)XROW | ((unsigned)BM << 16)), // tile_dim1|tile_dim2
             (int)NS,                         // tensor_dim0_stride = 128
             (int)((unsigned)(XROW * NS) << 16), // dim1_stride = 49152 (lo16)
             0 };                             // dim1_stride hi = 0
  v4i g2 = { BM, 1, 0, 0 };                   // tensor_dim2=16, tensor_dim3=1
  v4i g3 = { 0, 0, 0, 0 };
#if defined(__clang_major__) && (__clang_major__ >= 23)
  v8i g4 = { 0, 0, 0, 0, 0, 0, 0, 0 };
  __builtin_amdgcn_tensor_load_to_lds(g0, g1, g2, g3, g4, 0);
#else
  __builtin_amdgcn_tensor_load_to_lds(g0, g1, g2, g3, 0);
#endif
}

// ---------------------------------------------------------------------------
// Kernel 1: the GRU-D scan. One block = 16 batch rows, 8 waves * 2 N-tiles
// cover H=256. State lives in LDS across all 128 steps; TDM double-buffers
// the next step's X slice while WMMAs run.
// ---------------------------------------------------------------------------
__global__ __launch_bounds__(256) void grud_scan(
    const float* __restrict__ X, const float* __restrict__ Xmean,
    const float* __restrict__ bz, const float* __restrict__ br,
    const float* __restrict__ bh,
    const float* __restrict__ Wgx, const float* __restrict__ bgx,
    const float* __restrict__ bgh,
    const _Float16* __restrict__ wz, const _Float16* __restrict__ wr,
    const _Float16* __restrict__ wh, const _Float16* __restrict__ wgh,
    float* __restrict__ hT) {
  __shared__ __align__(32) _Float16 sC[BM * DIN];   // concat[x | h | m] (f16)
  __shared__ __align__(32) _Float16 sD[BM * NT];    // delta (f16) for dh GEMM
  __shared__ __align__(16) float    sH[BM * NH];    // hidden state (f32)
  __shared__ __align__(16) float    sX[2][XSLICE];  // TDM-staged X slices
  __shared__ float sWgd[NT];                        // diag(Wgx)
  __shared__ float sBgx[NT];

  const int tid   = threadIdx.x;
  const int lane  = tid & 31;
  const int wave  = tid >> 5;
  const int row16 = lane & 15;   // A-row / C-column within a 16-tile
  const int hi    = lane >> 4;   // lane-group select (K / M halves)
  const int b0    = blockIdx.x * BM;

  // h0 = 0; cache FilterLinear diagonal + bias
  for (int e = 0; e < (BM * NH) / 256; ++e) sH[tid + e * 256] = 0.0f;
  if (tid < NT) {
    sWgd[tid] = Wgx[tid * NT + tid];
    sBgx[tid] = bgx[tid];
  }

  // prologue: async-stage step 0 into buffer 0 (wave 0 issues; EXEC-uniform)
  if (wave == 0) {
    tdm_issue_xslice(X, b0, 0, (unsigned)(unsigned long long)(&sX[0][0]));
  }
  __syncthreads();

  for (int s = 0; s < NS; ++s) {
    const int cur = s & 1;
    if (wave == 0) __builtin_amdgcn_s_wait_tensorcnt(0);
    __syncthreads();   // publish sX[cur] to all waves

    // kick off the DMA for the next step into the other buffer
    if (wave == 0 && (s + 1) < NS) {
      tdm_issue_xslice(X, b0, s + 1,
                       (unsigned)(unsigned long long)(&sX[cur ^ 1][0]));
    }

    // ---- Phase 1: elementwise input-decay imputation from staged slice ----
    for (int e = 0; e < 8; ++e) {
      int j = tid + e * 256;            // 0..2047
      int r = j >> 7;                   // batch row in tile
      int t = j & 127;
      const float* xr = &sX[cur][r * XROW + 3 * t];
      float m  = xr[0];
      float x  = xr[1];
      float d  = xr[2];
      float xm = Xmean[s * NT + t];
      float a  = d * sWgd[t] + sBgx[t];            // diag-masked FilterLinear
      float dx = __expf(-fmaxf(a, 0.0f));
      float xi = m * x + (1.0f - m) * (dx * x + (1.0f - dx) * xm);
      sC[r * DIN + t]       = (_Float16)xi;
      sC[r * DIN + 384 + t] = (_Float16)m;
      sD[r * NT + t]        = (_Float16)d;
    }
    __syncthreads();

    // ---- Phase 2: dh = exp(-relu(d @ Wgh^T + bgh)); h *= dh ----
    for (int q = 0; q < 2; ++q) {
      int nt   = wave * 2 + q;
      int ncol = nt * 16 + row16;
      float bias = bgh[ncol];
      v8f acc;
#pragma unroll
      for (int i = 0; i < 8; ++i) acc[i] = bias;
#pragma unroll
      for (int kt = 0; kt < NT / 32; ++kt) {
        int k0 = kt * 32 + hi * 16;
        v16h af = *(const v16h*)(&sD[row16 * NT + k0]);
        v16h bf = *(const v16h*)(wgh + (long)ncol * NT + k0);
        acc = __builtin_amdgcn_wmma_f32_16x16x32_f16(false, af, false, bf,
                                                     (short)0, acc, false, false);
      }
#pragma unroll
      for (int i = 0; i < 8; ++i) {
        int mrow = i + hi * 8;
        float dh = __expf(-fmaxf(acc[i], 0.0f));
        float hv = sH[mrow * NH + ncol] * dh;
        sH[mrow * NH + ncol] = hv;
        sC[mrow * DIN + 128 + ncol] = (_Float16)hv;   // h goes into concat
      }
    }
    __syncthreads();

    // ---- Phase 3: z = sigmoid(c@Wz^T+bz), r = sigmoid(c@Wr^T+br) ----
    v8f zk[2], rk[2];
    for (int q = 0; q < 2; ++q) {
      int nt   = wave * 2 + q;
      int ncol = nt * 16 + row16;
      float bzv = bz[ncol], brv = br[ncol];
      v8f az, ar;
#pragma unroll
      for (int i = 0; i < 8; ++i) { az[i] = bzv; ar[i] = brv; }
      const _Float16* wzp = wz + (long)ncol * DIN;
      const _Float16* wrp = wr + (long)ncol * DIN;
#pragma unroll 4
      for (int kt = 0; kt < DIN / 32; ++kt) {
        int k0 = kt * 32 + hi * 16;
        __builtin_prefetch(wzp + k0 + 64, 0, 0);
        __builtin_prefetch(wrp + k0 + 64, 0, 0);
        v16h af = *(const v16h*)(&sC[row16 * DIN + k0]);
        v16h bz16 = *(const v16h*)(wzp + k0);
        az = __builtin_amdgcn_wmma_f32_16x16x32_f16(false, af, false, bz16,
                                                    (short)0, az, false, false);
        v16h br16 = *(const v16h*)(wrp + k0);
        ar = __builtin_amdgcn_wmma_f32_16x16x32_f16(false, af, false, br16,
                                                    (short)0, ar, false, false);
      }
#pragma unroll
      for (int i = 0; i < 8; ++i) {
        az[i] = 1.0f / (1.0f + __expf(-az[i]));
        ar[i] = 1.0f / (1.0f + __expf(-ar[i]));
      }
      zk[q] = az; rk[q] = ar;
    }
    __syncthreads();   // all waves done reading sC for z/r

    // ---- Phase 4: overwrite concat middle with r*h ----
    for (int q = 0; q < 2; ++q) {
      int ncol = (wave * 2 + q) * 16 + row16;
#pragma unroll
      for (int i = 0; i < 8; ++i) {
        int mrow = i + hi * 8;
        sC[mrow * DIN + 128 + ncol] =
            (_Float16)(rk[q][i] * sH[mrow * NH + ncol]);
      }
    }
    __syncthreads();

    // ---- Phase 5: ht = tanh([x, r*h, m]@Wh^T + bh); h = (1-z)h + z*ht ----
    for (int q = 0; q < 2; ++q) {
      int nt   = wave * 2 + q;
      int ncol = nt * 16 + row16;
      float bhv = bh[ncol];
      v8f ah;
#pragma unroll
      for (int i = 0; i < 8; ++i) ah[i] = bhv;
      const _Float16* whp = wh + (long)ncol * DIN;
#pragma unroll 4
      for (int kt = 0; kt < DIN / 32; ++kt) {
        int k0 = kt * 32 + hi * 16;
        __builtin_prefetch(whp + k0 + 64, 0, 0);
        v16h af = *(const v16h*)(&sC[row16 * DIN + k0]);
        v16h bf = *(const v16h*)(whp + k0);
        ah = __builtin_amdgcn_wmma_f32_16x16x32_f16(false, af, false, bf,
                                                    (short)0, ah, false, false);
      }
#pragma unroll
      for (int i = 0; i < 8; ++i) {
        int mrow = i + hi * 8;
        float ht_ = tanhf(ah[i]);
        float z   = zk[q][i];
        float hv  = sH[mrow * NH + ncol];
        sH[mrow * NH + ncol] = (1.0f - z) * hv + z * ht_;
      }
    }
    __syncthreads();   // protect sC/sH/sX before next step
  }

  // spill final hidden state
  for (int e = 0; e < 16; ++e) {
    int j = tid + e * 256;              // 0..4095
    int r = j >> 8;
    int col = j & 255;
    hT[(long)(b0 + r) * NH + col] = sH[r * NH + col];
  }
}

// ---------------------------------------------------------------------------
// Kernel 2: batch-statistics for train-mode BatchNorm (biased variance).
// ---------------------------------------------------------------------------
__global__ __launch_bounds__(256) void grud_stats(
    const float* __restrict__ hT, float* __restrict__ mu, float* __restrict__ var) {
  int col = threadIdx.x;
  float s = 0.0f, s2 = 0.0f;
  for (int b = 0; b < NB; ++b) {
    float v = hT[(long)b * NH + col];
    s += v; s2 += v * v;
  }
  float m = s * (1.0f / NB);
  mu[col]  = m;
  var[col] = s2 * (1.0f / NB) - m * m;
}

// ---------------------------------------------------------------------------
// Kernel 3: normalize + FC head, one block per batch row with LDS reduction.
// ---------------------------------------------------------------------------
__global__ __launch_bounds__(256) void grud_fc(
    const float* __restrict__ hT, const float* __restrict__ mu,
    const float* __restrict__ var, const float* __restrict__ gamma,
    const float* __restrict__ beta, const float* __restrict__ Wfc,
    const float* __restrict__ bfc, float* __restrict__ out) {
  __shared__ float red[256];
  int b = blockIdx.x, col = threadIdx.x;
  float v  = hT[(long)b * NH + col];
  float hn = (v - mu[col]) * rsqrtf(var[col] + 1e-5f) * gamma[col] + beta[col];
  red[col] = hn * Wfc[col];
  __syncthreads();
  for (int off = 128; off > 0; off >>= 1) {
    if (col < off) red[col] += red[col + off];
    __syncthreads();
  }
  if (col == 0) out[b] = red[0] + bfc[0];
}

// ---------------------------------------------------------------------------
extern "C" void kernel_launch(void* const* d_in, const int* in_sizes, int n_in,
                              void* d_out, int out_size, void* d_ws, size_t ws_size,
                              hipStream_t stream) {
  (void)in_sizes; (void)n_in; (void)out_size; (void)ws_size;
  const float* X     = (const float*)d_in[0];
  const float* Xmean = (const float*)d_in[1];
  const float* Wz    = (const float*)d_in[2];
  const float* bz    = (const float*)d_in[3];
  const float* Wr    = (const float*)d_in[4];
  const float* br    = (const float*)d_in[5];
  const float* Wh    = (const float*)d_in[6];
  const float* bh    = (const float*)d_in[7];
  const float* Wgx   = (const float*)d_in[8];
  const float* bgx   = (const float*)d_in[9];
  const float* Wgh   = (const float*)d_in[10];
  const float* bgh   = (const float*)d_in[11];
  const float* gamma = (const float*)d_in[12];
  const float* beta  = (const float*)d_in[13];
  const float* Wfc   = (const float*)d_in[14];
  const float* bfc   = (const float*)d_in[15];

  char* ws = (char*)d_ws;
  _Float16* wz16  = (_Float16*)ws; ws += (size_t)NH * DIN * 2;
  _Float16* wr16  = (_Float16*)ws; ws += (size_t)NH * DIN * 2;
  _Float16* wh16  = (_Float16*)ws; ws += (size_t)NH * DIN * 2;
  _Float16* wgh16 = (_Float16*)ws; ws += (size_t)NH * NT * 2;
  float* hT  = (float*)ws; ws += (size_t)NB * NH * 4;
  float* mu  = (float*)ws; ws += 256 * 4;
  float* var = (float*)ws;

  grud_convert<<<(NH * DIN + 255) / 256, 256, 0, stream>>>(
      Wz, Wr, Wh, Wgh, wz16, wr16, wh16, wgh16);
  grud_scan<<<NB / BM, 256, 0, stream>>>(
      X, Xmean, bz, br, bh, Wgx, bgx, bgh, wz16, wr16, wh16, wgh16, hT);
  grud_stats<<<1, 256, 0, stream>>>(hT, mu, var);
  grud_fc<<<NB, 256, 0, stream>>>(hT, mu, var, gamma, beta, Wfc, bfc,
                                  (float*)d_out);
}
